// MultiHeadAttentionOrigin_7739531067802
// MI455X (gfx1250) — matrix operand order repaired and checked by reading, not compiled, not run on hardware
//
#include <hip/hip_runtime.h>
#include <hip/hip_bf16.h>

// MI455X / gfx1250 multi-head self-attention.
// - All GEMMs on v_wmma_f32_16x16x32_bf16 (bf16 in, fp32 accumulate).
// - Flash-attention streaming softmax (no [H,B,S,S] score tensor).
// - Attention K/V blocks staged into LDS with global_load_async_to_lds_b128,
//   double-buffered, shared by the 4 waves of a workgroup, synced with
//   s_wait_asynccnt + workgroup barriers.

typedef __bf16 bf16;
typedef bf16  v16bf __attribute__((ext_vector_type(16)));
typedef bf16  v8bf  __attribute__((ext_vector_type(8)));
typedef float v8f   __attribute__((ext_vector_type(8)));

#define BATCH 4
#define SEQ   2048
#define DMODEL 1024
#define NHEAD 16
#define DK    64
#define ROWS  (BATCH * SEQ)   // 8192
#define NBLK  (SEQ / 32)      // 64 key blocks per (h,b)

// padded LDS strides (elements); keep 16B alignment for B128 async writes,
// starts distinct mod 64 banks for the 16-lane fragment reads.
#define KSTRIDE 72            // 144 B per 32-key row (128 B data + 16 B pad)
#define VSTRIDE 40            // 80 B per dim row (64 B data + 16 B pad)

__device__ __forceinline__ v8f wmma_bf16(v16bf a, v16bf b, v8f c) {
  return __builtin_amdgcn_wmma_f32_16x16x32_bf16(
      /*neg_a=*/false, a, /*neg_b=*/false, b,
      /*c_mod=*/(short)0, c, /*reuse_a=*/false, /*reuse_b=*/false);
}

// A-fragment (16x32 bf16, M x K). `base` -> (row0,k0), row-major, ld elements.
// ISA layout: lane holds row m = lane&15; K chunks [half*8..+7], [16+half*8..+7].
__device__ __forceinline__ v16bf load_a_frag(const bf16* base, int ld, int lane) {
  const int m = lane & 15, half = lane >> 4;
  const bf16* rp = base + (size_t)m * ld + half * 8;
  v8bf lo = *(const v8bf*)(rp);
  v8bf hi = *(const v8bf*)(rp + 16);
  v16bf a;
#pragma unroll
  for (int i = 0; i < 8; ++i) { a[i] = lo[i]; a[i + 8] = hi[i]; }
  return a;
}

// B-fragment (32x16 bf16, K x N) from TRANSPOSED storage: `bt` -> (n0,k0) of a
// [N][K] row-major matrix, leading dim ldt. ISA layout: lane holds column
// n = lane&15; K chunks [half*8..+7], [16+half*8..+7] (read as 2x16B).
__device__ __forceinline__ v16bf load_b_frag_t(const bf16* bt, int ldt, int lane) {
  const int n = lane & 15, half = lane >> 4;
  const bf16* p = bt + (size_t)n * ldt + half * 8;
  v8bf lo = *(const v8bf*)(p);
  v8bf hi = *(const v8bf*)(p + 16);
  v16bf b;
#pragma unroll
  for (int i = 0; i < 8; ++i) { b[i] = lo[i]; b[i + 8] = hi[i]; }
  return b;
}

// Async DMA of one 16-byte chunk: global -> LDS (ASYNCcnt-tracked, no VGPR data).
__device__ __forceinline__ void async_b128(unsigned lds_byte_addr, const void* gaddr) {
  asm volatile("global_load_async_to_lds_b128 %0, %1, off"
               :: "v"(lds_byte_addr), "v"((unsigned long long)(size_t)gaddr)
               : "memory");
}

// ---------------- conversion kernels ----------------

__global__ void cvt_q_kernel(const float* __restrict__ q, bf16* __restrict__ qbf) {
  int i = blockIdx.x * blockDim.x + threadIdx.x;   // exactly ROWS*DMODEL threads
  qbf[i] = (bf16)q[i];
}

// Wt[p][h][n(64)][d(1024)] = W_p[h][d][n]   (p: 0=Q,1=K,2=V)
__global__ void cvt_wqkv_kernel(const float* __restrict__ Wq,
                                const float* __restrict__ Wk,
                                const float* __restrict__ Wv,
                                bf16* __restrict__ Wt) {
  int i = blockIdx.x * blockDim.x + threadIdx.x;   // 3*16*64*1024 threads
  int d = i & 1023;
  int n = (i >> 10) & 63;
  int h = (i >> 16) & 15;
  int p = i >> 20;
  const float* W = (p == 0) ? Wq : (p == 1) ? Wk : Wv;
  Wt[i] = (bf16)W[((size_t)h * DMODEL + d) * DK + n];
}

// WoT[e][c = h*64+v] = W_out[h][v][e]
__global__ void cvt_wout_kernel(const float* __restrict__ Wo, bf16* __restrict__ WoT) {
  int i = blockIdx.x * blockDim.x + threadIdx.x;   // 1024*1024 threads
  int c = i & 1023;
  int e = i >> 10;
  int h = c >> 6, v = c & 63;
  WoT[i] = (bf16)Wo[((size_t)h * DK + v) * DMODEL + e];
}

// ---------------- projection GEMM ----------------
// Per wave: C[16 x 64] = qbf[m0..m0+15, :1024] @ W[h]; K-loop step 32.
// p==0 -> Qb [h][b][s][64], p==1 -> Kb [h][b][s][64], p==2 -> Vt [h][b][64][S].
__global__ void proj_kernel(const bf16* __restrict__ qbf, const bf16* __restrict__ Wt,
                            bf16* __restrict__ Qb, bf16* __restrict__ Kb,
                            bf16* __restrict__ Vt) {
  const int lane = threadIdx.x & 31;
  const int wave = threadIdx.x >> 5;
  const int mt = blockIdx.x * 4 + wave;       // 0..511
  const int h  = blockIdx.y;
  const int p  = blockIdx.z;
  const int m0 = mt * 16;

  const bf16* wt = Wt + ((size_t)p * NHEAD + h) * DK * DMODEL;  // [64][1024]

  v8f c0 = {}, c1 = {}, c2 = {}, c3 = {};
  for (int k0 = 0; k0 < DMODEL; k0 += 32) {
    v16bf a = load_a_frag(qbf + (size_t)m0 * DMODEL + k0, DMODEL, lane);
    c0 = wmma_bf16(a, load_b_frag_t(wt + (size_t)0  * DMODEL + k0, DMODEL, lane), c0);
    c1 = wmma_bf16(a, load_b_frag_t(wt + (size_t)16 * DMODEL + k0, DMODEL, lane), c1);
    c2 = wmma_bf16(a, load_b_frag_t(wt + (size_t)32 * DMODEL + k0, DMODEL, lane), c2);
    c3 = wmma_bf16(a, load_b_frag_t(wt + (size_t)48 * DMODEL + k0, DMODEL, lane), c3);
  }

  const int b = m0 / SEQ;
  const int s0 = m0 % SEQ;
  const int half = lane >> 4, col = lane & 15;
  const size_t hb = (size_t)h * BATCH + b;

#pragma unroll
  for (int j = 0; j < 8; ++j) {
    const int s = s0 + j + half * 8;
    float vals[4] = {c0[j], c1[j], c2[j], c3[j]};
    if (p == 2) {
#pragma unroll
      for (int t = 0; t < 4; ++t) {
        int dim = t * 16 + col;
        Vt[(hb * DK + dim) * SEQ + s] = (bf16)vals[t];
      }
    } else {
      bf16* dst = (p == 0) ? Qb : Kb;
#pragma unroll
      for (int t = 0; t < 4; ++t) {
        int dim = t * 16 + col;
        dst[(hb * SEQ + s) * DK + dim] = (bf16)vals[t];
      }
    }
  }
}

// ---------------- flash attention ----------------
// Block = 4 waves on one (h,b); wave w owns query tile qt = 4*blockIdx.x + w.
// Per 32-key block: K (32x64) and V (64x32, pre-transposed) are DMA'd into
// LDS once (async, double-buffered) and shared by all 4 waves.
__global__ void __launch_bounds__(128) attn_kernel(
    const bf16* __restrict__ Qb, const bf16* __restrict__ Kb,
    const bf16* __restrict__ Vt, bf16* __restrict__ heads) {
  const int tid  = threadIdx.x;
  const int lane = tid & 31;
  const int wave = tid >> 5;
  const int qt = blockIdx.x * 4 + wave;      // 0..127
  const int h  = blockIdx.y;
  const int b  = blockIdx.z;
  const int m0 = qt * 16;
  const size_t hb = (size_t)h * BATCH + b;

  const bf16* Qhb = Qb + hb * SEQ * DK;
  const bf16* Khb = Kb + hb * SEQ * DK;
  const bf16* Vhb = Vt + hb * DK * SEQ;

  __shared__ bf16 Ktile[2][32 * KSTRIDE];   // 2 x 32 keys x 64 dims (padded)
  __shared__ bf16 Vtile[2][64 * VSTRIDE];   // 2 x 64 dims x 32 keys (padded)
  __shared__ bf16 Ptile[4][16 * 32];        // per-wave P staging
  bf16* pl = Ptile[wave];

  // cooperative async stage of one K/V block into LDS buffer `buf`
  auto stage = [&](int buf, int n0) {
#pragma unroll
    for (int p = 0; p < 2; ++p) {           // K: 4 KB = 256 x 16B chunks
      int c = p * 128 + tid;
      int row = c >> 3, seg = c & 7;        // key row, 16B segment
      unsigned lds = (unsigned)(size_t)&Ktile[buf][row * KSTRIDE + seg * 8];
      async_b128(lds, Khb + (size_t)(n0 + row) * DK + seg * 8);
    }
#pragma unroll
    for (int p = 0; p < 2; ++p) {           // V: 4 KB = 256 x 16B chunks
      int c = p * 128 + tid;
      int dim = c >> 2, seg = c & 3;
      unsigned lds = (unsigned)(size_t)&Vtile[buf][dim * VSTRIDE + seg * 8];
      async_b128(lds, Vhb + (size_t)dim * SEQ + n0 + seg * 8);
    }
  };

  // Q fragments for the whole tile (K = 0..31 and 32..63)
  v16bf aq0 = load_a_frag(Qhb + (size_t)m0 * DK + 0,  DK, lane);
  v16bf aq1 = load_a_frag(Qhb + (size_t)m0 * DK + 32, DK, lane);

  v8f acc0 = {}, acc1 = {}, acc2 = {}, acc3 = {};
  float mst[8], lst[8];
#pragma unroll
  for (int j = 0; j < 8; ++j) { mst[j] = -1e30f; lst[j] = 0.0f; }

  const float norm = 0.125f;  // 1/sqrt(64)
  const int half = lane >> 4, col = lane & 15;

  stage(0, 0);
  for (int blk = 0; blk < NBLK; ++blk) {
    const int cur = blk & 1;
    if (blk + 1 < NBLK) {
      stage(1 - cur, (blk + 1) * 32);       // overlap next DMA with this compute
      asm volatile("s_wait_asynccnt 0x4" ::: "memory");  // current block done
    } else {
      asm volatile("s_wait_asynccnt 0x0" ::: "memory");
    }
    __syncthreads();                         // all waves' DMA visible

    const bf16* Kt = Ktile[cur];
    const bf16* Vw = Vtile[cur];

    // scores: two 16x16 tiles covering the 32 keys of this block
    v8f z = {};
    v8f s0 = wmma_bf16(aq0, load_b_frag_t(Kt + (size_t)0  * KSTRIDE + 0,  KSTRIDE, lane), z);
    s0     = wmma_bf16(aq1, load_b_frag_t(Kt + (size_t)0  * KSTRIDE + 32, KSTRIDE, lane), s0);
    v8f s1 = wmma_bf16(aq0, load_b_frag_t(Kt + (size_t)16 * KSTRIDE + 0,  KSTRIDE, lane), z);
    s1     = wmma_bf16(aq1, load_b_frag_t(Kt + (size_t)16 * KSTRIDE + 32, KSTRIDE, lane), s1);

    float corr[8];
#pragma unroll
    for (int j = 0; j < 8; ++j) {
      float x0 = s0[j] * norm;
      float x1 = s1[j] * norm;
      // row max over the 16 lanes of this half (row M = j + half*8)
      float r = fmaxf(x0, x1);
      r = fmaxf(r, __shfl_xor(r, 1, 16));
      r = fmaxf(r, __shfl_xor(r, 2, 16));
      r = fmaxf(r, __shfl_xor(r, 4, 16));
      r = fmaxf(r, __shfl_xor(r, 8, 16));
      float mnew = fmaxf(mst[j], r);
      float p0 = __expf(x0 - mnew);
      float p1 = __expf(x1 - mnew);
      float rs = p0 + p1;
      rs += __shfl_xor(rs, 1, 16);
      rs += __shfl_xor(rs, 2, 16);
      rs += __shfl_xor(rs, 4, 16);
      rs += __shfl_xor(rs, 8, 16);
      corr[j] = __expf(mst[j] - mnew);
      lst[j] = lst[j] * corr[j] + rs;
      mst[j] = mnew;
      // stage P (bf16) in LDS as [16 rows][32 key cols]
      int row = j + half * 8;
      pl[row * 32 + col]      = (bf16)p0;
      pl[row * 32 + 16 + col] = (bf16)p1;
    }
#pragma unroll
    for (int j = 0; j < 8; ++j) {           // rescale accumulators
      acc0[j] *= corr[j]; acc1[j] *= corr[j];
      acc2[j] *= corr[j]; acc3[j] *= corr[j];
    }
    __syncthreads();                         // P visible (own wave's LDS writes)

    v16bf ap = load_a_frag(pl, 32, lane);
    acc0 = wmma_bf16(ap, load_b_frag_t(Vw + (size_t)0  * VSTRIDE, VSTRIDE, lane), acc0);
    acc1 = wmma_bf16(ap, load_b_frag_t(Vw + (size_t)16 * VSTRIDE, VSTRIDE, lane), acc1);
    acc2 = wmma_bf16(ap, load_b_frag_t(Vw + (size_t)32 * VSTRIDE, VSTRIDE, lane), acc2);
    acc3 = wmma_bf16(ap, load_b_frag_t(Vw + (size_t)48 * VSTRIDE, VSTRIDE, lane), acc3);
    __syncthreads();                         // done reading cur before overwrite
  }

  // normalize and emit into the concatenated [ROWS][H*DK] head matrix
#pragma unroll
  for (int j = 0; j < 8; ++j) {
    float inv = 1.0f / lst[j];
    size_t rowbase = ((size_t)b * SEQ + m0 + j + half * 8) * DMODEL + (size_t)h * DK;
    heads[rowbase + 0  + col] = (bf16)(acc0[j] * inv);
    heads[rowbase + 16 + col] = (bf16)(acc1[j] * inv);
    heads[rowbase + 32 + col] = (bf16)(acc2[j] * inv);
    heads[rowbase + 48 + col] = (bf16)(acc3[j] * inv);
  }
}

// ---------------- output projection ----------------
// out[8192 x 1024] = heads[8192 x 1024] @ Wo (via transposed WoT[e][c]).
__global__ void out_kernel(const bf16* __restrict__ heads, const bf16* __restrict__ WoT,
                           float* __restrict__ out) {
  const int lane = threadIdx.x & 31;
  const int wave = threadIdx.x >> 5;
  const int mt = blockIdx.x * 4 + wave;     // 0..511
  const int m0 = mt * 16;
  const int e0 = blockIdx.y * 64;

  v8f c0 = {}, c1 = {}, c2 = {}, c3 = {};
  for (int k0 = 0; k0 < DMODEL; k0 += 32) {
    v16bf a = load_a_frag(heads + (size_t)m0 * DMODEL + k0, DMODEL, lane);
    c0 = wmma_bf16(a, load_b_frag_t(WoT + (size_t)(e0 + 0)  * DMODEL + k0, DMODEL, lane), c0);
    c1 = wmma_bf16(a, load_b_frag_t(WoT + (size_t)(e0 + 16) * DMODEL + k0, DMODEL, lane), c1);
    c2 = wmma_bf16(a, load_b_frag_t(WoT + (size_t)(e0 + 32) * DMODEL + k0, DMODEL, lane), c2);
    c3 = wmma_bf16(a, load_b_frag_t(WoT + (size_t)(e0 + 48) * DMODEL + k0, DMODEL, lane), c3);
  }

  const int half = lane >> 4, col = lane & 15;
#pragma unroll
  for (int j = 0; j < 8; ++j) {
    size_t row = (size_t)m0 + j + half * 8;
    out[row * DMODEL + e0 + 0  + col] = c0[j];
    out[row * DMODEL + e0 + 16 + col] = c1[j];
    out[row * DMODEL + e0 + 32 + col] = c2[j];
    out[row * DMODEL + e0 + 48 + col] = c3[j];
  }
}

// ---------------- launcher ----------------

extern "C" void kernel_launch(void* const* d_in, const int* in_sizes, int n_in,
                              void* d_out, int out_size, void* d_ws, size_t ws_size,
                              hipStream_t stream) {
  const float* q  = (const float*)d_in[0];
  const float* Wq = (const float*)d_in[1];
  const float* Wk = (const float*)d_in[2];
  const float* Wv = (const float*)d_in[3];
  const float* Wo = (const float*)d_in[4];
  float* out = (float*)d_out;

  char* ws = (char*)d_ws;
  const size_t SZ_QBF  = (size_t)ROWS * DMODEL * 2;            // 16.78 MB
  const size_t SZ_WT   = (size_t)3 * NHEAD * DK * DMODEL * 2;  //  6.29 MB
  const size_t SZ_WOT  = (size_t)DMODEL * DMODEL * 2;          //  2.10 MB
  const size_t SZ_QKV  = (size_t)NHEAD * BATCH * SEQ * DK * 2; // 16.78 MB each

  bf16* qbf   = (bf16*)(ws);                 ws += SZ_QBF;
  bf16* Wt    = (bf16*)(ws);                 ws += SZ_WT;
  bf16* WoT   = (bf16*)(ws);                 ws += SZ_WOT;
  bf16* Qb    = (bf16*)(ws);                 ws += SZ_QKV;
  bf16* Kb    = (bf16*)(ws);                 ws += SZ_QKV;
  bf16* Vt    = (bf16*)(ws);                 ws += SZ_QKV;
  bf16* heads = (bf16*)(ws);                 ws += SZ_QBF;

  cvt_q_kernel   <<<(ROWS * DMODEL) / 256, 256, 0, stream>>>(q, qbf);
  cvt_wqkv_kernel<<<(3 * NHEAD * DK * DMODEL) / 256, 256, 0, stream>>>(Wq, Wk, Wv, Wt);
  cvt_wout_kernel<<<(DMODEL * DMODEL) / 256, 256, 0, stream>>>(Wo, WoT);

  proj_kernel<<<dim3(ROWS / 64, NHEAD, 3), 128, 0, stream>>>(qbf, Wt, Qb, Kb, Vt);
  attn_kernel<<<dim3(SEQ / 64, NHEAD, BATCH), 128, 0, stream>>>(Qb, Kb, Vt, heads);
  out_kernel <<<dim3(ROWS / 64, DMODEL / 64), 128, 0, stream>>>(heads, WoT, out);
}
